// Decoder_54082228191615
// MI455X (gfx1250) — compile-verified, matrix operand order checked by reading, not run on hardware
//
#include <hip/hip_runtime.h>

// ---------------------------------------------------------------------------
// 2-layer GPT-style decoder forward for MI455X (gfx1250, wave32, WMMA).
// All GEMMs: fp16 operands (pre-converted, B pre-transposed to [n][k]),
// fp32 accumulate via v_wmma_f32_16x16x32_f16, 128x64 block tile,
// 4 WMMAs per wave per K-step. LDS staging uses GLOBAL_LOAD_ASYNC_TO_LDS_B128
// (signature discovered via compile probe: (global v4i*, local v4i*, imm, imm)).
// Workspace: ~85 MB. d_out = [ x : B*S*D f32 ][ attns : L*B*H*S*S f32 ]
// ---------------------------------------------------------------------------

typedef _Float16 v16h __attribute__((ext_vector_type(16)));
typedef _Float16 h8   __attribute__((ext_vector_type(8)));
typedef float    v8f  __attribute__((ext_vector_type(8)));
typedef int      v4i  __attribute__((vector_size(16)));

#define DMODEL 768
#define NHEADS 12
#define DK     64
#define DFF    3072
#define BB     2
#define SS     2048
#define NLAY   2

// ---- gfx1250 async-to-LDS staging (guarded; falls back to sync copy) ------
#if defined(__has_builtin)
#if __has_builtin(__builtin_amdgcn_global_load_async_to_lds_b128)
#define HAVE_ASYNC_LDS 1
#endif
#endif
#ifndef HAVE_ASYNC_LDS
#define HAVE_ASYNC_LDS 0
#endif

#define GLOBAL_AS __attribute__((address_space(1)))
#define LDS_AS    __attribute__((address_space(3)))

__device__ __forceinline__ void stage16(const _Float16* g, _Float16* l) {
#if HAVE_ASYNC_LDS
    __builtin_amdgcn_global_load_async_to_lds_b128(
        (GLOBAL_AS v4i*)g, (LDS_AS v4i*)l, 0, 0);
#else
    *(h8*)l = *(const h8*)g;
#endif
}

__device__ __forceinline__ void stage_wait() {
#if HAVE_ASYNC_LDS
#if __has_builtin(__builtin_amdgcn_s_wait_asynccnt)
    __builtin_amdgcn_s_wait_asynccnt(0);
#else
    asm volatile("s_wait_asynccnt 0x0" ::: "memory");
#endif
#endif
}

// Fragment load per documented 16-bit 16x32 layout: lane group g selects
// K-halves {8g..8g+7, 16+8g..23+8g}: two aligned 16-byte LDS reads.
__device__ __forceinline__ v16h load_frag(const _Float16* rowp, int g) {
    h8 lo = *(const h8*)(rowp + 8 * g);
    h8 hi = *(const h8*)(rowp + 16 + 8 * g);
    return __builtin_shufflevector(lo, hi,
             0,1,2,3,4,5,6,7,8,9,10,11,12,13,14,15);
}

// ---------------------------------------------------------------------------
// Strided-batched WMMA GEMM:  C[z] = act( A[z] (MxK) * Bt[z] + bias )
//   A: f16 (A_F32=0) or f32 (A_F32=1), element (m,k) at baseA + m*sArow + k
//      (k always contiguous).
//   B: f16, PRE-TRANSPOSED [n][k]: element (k,n) at baseB + n*sBrow + k.
//   C: f32 (C32) or f16 (C16) at baseC + m*sCm + n*sCn.
// Block 256 threads = 8 waves, tile 128x64, K-step 32. Wave w: rows
// (w>>1)*32..+31, cols (w&1)*32..+31 -> 4 accumulators, 4 WMMAs / K-step.
// Requires M%128==0, N%64==0, K%32==0 (true for all calls): no divergence.
// ---------------------------------------------------------------------------
template<int A_F32>
__global__ __launch_bounds__(256) void gemm_wmma(
    const void* __restrict__ Av, long long sAb0, long long sAb1, int sArow,
    const _Float16* __restrict__ Bp, long long sBb0, long long sBb1, int sBrow,
    float* __restrict__ C32, _Float16* __restrict__ C16,
    long long sCb0, long long sCb1, int sCm, int sCn,
    int K, int Hdim, const float* __restrict__ bias, int relu)
{
    __shared__ __align__(16) _Float16 As[128][32];
    __shared__ __align__(16) _Float16 Bs[64][32];

    const int tid  = threadIdx.x;
    const int lane = tid & 31;
    const int wave = tid >> 5;
    const int wr   = wave >> 1;   // 0..3 : 32-row strip
    const int wc   = wave & 1;    // 0..1 : 32-col strip
    const int g    = lane >> 4;
    const int l16  = lane & 15;

    const int zb = blockIdx.z / Hdim;
    const int zh = blockIdx.z % Hdim;
    const long long baseA = (long long)zb * sAb0 + (long long)zh * sAb1;
    const long long baseB = (long long)zb * sBb0 + (long long)zh * sBb1;
    const long long baseC = (long long)zb * sCb0 + (long long)zh * sCb1;

    const int m0 = blockIdx.y * 128;
    const int n0 = blockIdx.x * 64;

    v8f acc[4];
    #pragma unroll
    for (int a = 0; a < 4; ++a)
        #pragma unroll
        for (int v = 0; v < 8; ++v) acc[a][v] = 0.f;

    // B fill coordinates (one 16B chunk per thread per K-step)
    const int bn = tid >> 2;            // 0..63
    const int bc8 = (tid & 3) * 8;      // 0,8,16,24

    for (int k0 = 0; k0 < K; k0 += 32) {
        // ---- stage A tile (128x32 f16 = 8 KiB) ----
        if constexpr (A_F32) {
            const float* A32 = (const float*)Av;
            #pragma unroll
            for (int i = 0; i < 2; ++i) {
                int e = tid + i * 256;          // 0..511
                int r = e >> 2, c8 = (e & 3) * 8;
                const float* src = A32 + baseA + (long long)(m0 + r) * sArow
                                       + (k0 + c8);
                float4 f0 = *(const float4*)src;
                float4 f1 = *(const float4*)(src + 4);
                h8 h;
                h[0] = (_Float16)f0.x; h[1] = (_Float16)f0.y;
                h[2] = (_Float16)f0.z; h[3] = (_Float16)f0.w;
                h[4] = (_Float16)f1.x; h[5] = (_Float16)f1.y;
                h[6] = (_Float16)f1.z; h[7] = (_Float16)f1.w;
                *(h8*)&As[r][c8] = h;
            }
        } else {
            const _Float16* A16 = (const _Float16*)Av;
            #pragma unroll
            for (int i = 0; i < 2; ++i) {
                int e = tid + i * 256;
                int r = e >> 2, c8 = (e & 3) * 8;
                stage16(A16 + baseA + (long long)(m0 + r) * sArow + (k0 + c8),
                        &As[r][c8]);
            }
        }
        // ---- stage B tile (64x32 f16 = 4 KiB), already [n][k] in memory ----
        const _Float16* bsrc = Bp + baseB + (long long)(n0 + bn) * sBrow + k0 + bc8;
        stage16(bsrc, &Bs[bn][bc8]);
        if (k0 + 32 < K) __builtin_prefetch((const void*)(bsrc + 32), 0, 1);
        stage_wait();
        __syncthreads();

        // ---- fragments + 4 WMMAs ----
        v16h bf0 = load_frag(&Bs[wc * 32 + l16][0], g);
        v16h bf1 = load_frag(&Bs[wc * 32 + 16 + l16][0], g);
        v16h af0 = load_frag(&As[wr * 32 + l16][0], g);
        v16h af1 = load_frag(&As[wr * 32 + 16 + l16][0], g);

        acc[0] = __builtin_amdgcn_wmma_f32_16x16x32_f16(
                     false, af0, false, bf0, (short)0, acc[0], false, false);
        acc[1] = __builtin_amdgcn_wmma_f32_16x16x32_f16(
                     false, af0, false, bf1, (short)0, acc[1], false, false);
        acc[2] = __builtin_amdgcn_wmma_f32_16x16x32_f16(
                     false, af1, false, bf0, (short)0, acc[2], false, false);
        acc[3] = __builtin_amdgcn_wmma_f32_16x16x32_f16(
                     false, af1, false, bf1, (short)0, acc[3], false, false);
        __syncthreads();
    }

    // ---- epilogue: C layout = lane gives N (lane&15), VGPR v gives M=v+8g
    #pragma unroll
    for (int a = 0; a < 4; ++a) {
        const int nn = n0 + wc * 32 + (a & 1) * 16 + l16;
        const float bv = bias ? bias[nn] : 0.f;
        #pragma unroll
        for (int v = 0; v < 8; ++v) {
            int mm = m0 + wr * 32 + (a >> 1) * 16 + v + 8 * g;
            float val = acc[a][v] + bv;
            if (relu) val = fmaxf(val, 0.f);
            long long ci = baseC + (long long)mm * sCm + (long long)nn * sCn;
            if (C32) C32[ci] = val;
            else     C16[ci] = (_Float16)val;
        }
    }
}

// ---------------------------------------------------------------------------
// Weight prep: dst[n*K + k] = (f16)src[k*N + n]   (transpose + convert)
// ---------------------------------------------------------------------------
__global__ __launch_bounds__(256) void transpose_cvt(
    const float* __restrict__ src, _Float16* __restrict__ dst, int K, int N)
{
    __shared__ float tile[32][33];
    const int nb = blockIdx.x * 32, kb = blockIdx.y * 32;
    const int tx = threadIdx.x & 31, ty = threadIdx.x >> 5;   // 32 x 8
    #pragma unroll
    for (int j = 0; j < 4; ++j) {
        int ky = ty + j * 8;
        tile[ky][tx] = src[(long long)(kb + ky) * N + (nb + tx)];
    }
    __syncthreads();
    #pragma unroll
    for (int j = 0; j < 4; ++j) {
        int ny = ty + j * 8;
        dst[(long long)(nb + ny) * K + (kb + tx)] = (_Float16)tile[tx][ny];
    }
}

// ---------------------------------------------------------------------------
// In-place masked softmax over one attention row of length S.
// Faithful to reference: where(mask, scores, NEG) with mask = pad | future,
// i.e. KEEP the score when (attention_mask[b,k]==0) || (k > q), else -1e9.
// ---------------------------------------------------------------------------
__global__ __launch_bounds__(256) void softmax_mask(
    float* __restrict__ attn, const int* __restrict__ amask, float scale)
{
    const int q = blockIdx.x, h = blockIdx.y, b = blockIdx.z;
    float* row = attn + ((((long long)b * NHEADS + h) * SS) + q) * (long long)SS;
    const int tid = threadIdx.x;

    float v[8];
    float mx = -3.4e38f;
    #pragma unroll
    for (int j = 0; j < 8; ++j) {
        int kidx = tid + j * 256;
        bool keep = (amask[b * SS + kidx] == 0) || (kidx > q);
        float s = keep ? row[kidx] * scale : -1.0e9f;
        v[j] = s;
        mx = fmaxf(mx, s);
    }
    __shared__ float red[256];
    red[tid] = mx; __syncthreads();
    for (int s = 128; s > 0; s >>= 1) {
        if (tid < s) red[tid] = fmaxf(red[tid], red[tid + s]);
        __syncthreads();
    }
    mx = red[0]; __syncthreads();

    float sum = 0.f;
    #pragma unroll
    for (int j = 0; j < 8; ++j) { v[j] = __expf(v[j] - mx); sum += v[j]; }
    red[tid] = sum; __syncthreads();
    for (int s = 128; s > 0; s >>= 1) {
        if (tid < s) red[tid] += red[tid + s];
        __syncthreads();
    }
    const float inv = 1.f / red[0];
    #pragma unroll
    for (int j = 0; j < 8; ++j) row[tid + j * 256] = v[j] * inv;
}

// ---------------------------------------------------------------------------
// x = LayerNorm(inp + x) * g + b ; also refresh f16 shadow x16.
// ---------------------------------------------------------------------------
__global__ __launch_bounds__(256) void add_ln(
    const float* __restrict__ inp, float* __restrict__ x,
    _Float16* __restrict__ x16,
    const float* __restrict__ gg, const float* __restrict__ bb)
{
    const long long row = blockIdx.x;
    const int tid = threadIdx.x;
    float v[3];
    #pragma unroll
    for (int j = 0; j < 3; ++j) {
        int d = tid + j * 256;
        v[j] = inp[row * DMODEL + d] + x[row * DMODEL + d];
    }
    __shared__ float red[256];
    red[tid] = v[0] + v[1] + v[2]; __syncthreads();
    for (int s = 128; s > 0; s >>= 1) {
        if (tid < s) red[tid] += red[tid + s];
        __syncthreads();
    }
    const float mean = red[0] / (float)DMODEL; __syncthreads();

    float var = 0.f;
    #pragma unroll
    for (int j = 0; j < 3; ++j) { float d = v[j] - mean; var += d * d; }
    red[tid] = var; __syncthreads();
    for (int s = 128; s > 0; s >>= 1) {
        if (tid < s) red[tid] += red[tid + s];
        __syncthreads();
    }
    const float rstd = rsqrtf(red[0] / (float)DMODEL + 1e-5f);
    #pragma unroll
    for (int j = 0; j < 3; ++j) {
        int d = tid + j * 256;
        float o = (v[j] - mean) * rstd * gg[d] + bb[d];
        x[row * DMODEL + d] = o;
        x16[row * DMODEL + d] = (_Float16)o;
    }
}

// ---------------------------------------------------------------------------
__global__ void embed_kernel(const int* __restrict__ tok,
                             const float* __restrict__ emb,
                             const float* __restrict__ pos,
                             float* __restrict__ x, _Float16* __restrict__ x16)
{
    long long i = (long long)blockIdx.x * blockDim.x + threadIdx.x;
    if (i >= (long long)BB * SS * DMODEL) return;
    int d = (int)(i % DMODEL);
    long long rem = i / DMODEL;
    int s = (int)(rem % SS);
    int b = (int)(rem / SS);
    int t = tok[b * SS + s];
    float o = emb[(long long)t * DMODEL + d] + pos[(long long)s * DMODEL + d];
    x[i] = o;
    x16[i] = (_Float16)o;
}

__global__ void copy_kernel(const float* __restrict__ src,
                            float* __restrict__ dst, long long n)
{
    long long i = (long long)blockIdx.x * blockDim.x + threadIdx.x;
    if (i < n) dst[i] = src[i];
}

// ---------------------------------------------------------------------------
extern "C" void kernel_launch(void* const* d_in, const int* in_sizes, int n_in,
                              void* d_out, int out_size, void* d_ws, size_t ws_size,
                              hipStream_t stream)
{
    const int*   inputs = (const int*)  d_in[0];
    const int*   amask  = (const int*)  d_in[1];
    const float* emb    = (const float*)d_in[2];
    const float* pos    = (const float*)d_in[3];
    const float* wq     = (const float*)d_in[4];
    const float* wk     = (const float*)d_in[5];
    const float* wv     = (const float*)d_in[6];
    const float* wo     = (const float*)d_in[7];
    const float* ln1g   = (const float*)d_in[8];
    const float* ln1b   = (const float*)d_in[9];
    const float* w1     = (const float*)d_in[10];
    const float* b1     = (const float*)d_in[11];
    const float* w2     = (const float*)d_in[12];
    const float* b2     = (const float*)d_in[13];
    const float* ln2g   = (const float*)d_in[14];
    const float* ln2b   = (const float*)d_in[15];

    const long long ND     = (long long)BB * SS * DMODEL;          // 3,145,728
    const long long ATTN_L = (long long)BB * NHEADS * SS * SS;     // 100,663,296
    const long long WDD    = (long long)DMODEL * DMODEL;           // 589,824
    const long long WDF    = (long long)DMODEL * DFF;              // 2,359,296
    const long long LW     = 4 * WDD + 2 * WDF;                    // 7,077,888

    // ---- workspace layout (bytes): x | tmp | x16 | pool16 | wt16  (~85 MB)
    char* wsb = (char*)d_ws;
    float*    x      = (float*)   (wsb);
    float*    tmp    = (float*)   (wsb + ND * 4);
    _Float16* x16    = (_Float16*)(wsb + 2 * ND * 4);
    _Float16* pool16 = (_Float16*)(wsb + 2 * ND * 4 + ND * 2);
    _Float16* wt     = (_Float16*)(wsb + 2 * ND * 4 + ND * 2 + 4 * ND * 2);

    _Float16* q16  = pool16;
    _Float16* k16  = pool16 + ND;
    _Float16* v16t = pool16 + 2 * ND;   // [B,H,DK,S] (k-contiguous for ctx B)
    _Float16* c16  = pool16 + 3 * ND;   // ctx, [B,S,H*DK]
    _Float16* ffh16 = pool16;           // reuses q/k/v/ctx after attention

    float* out_x = (float*)d_out;
    float* attns = out_x + ND;

    const dim3 blk(256);

    // ---- one-time weight transpose+convert to f16 [n][k] ----
    for (int l = 0; l < NLAY; ++l) {
        _Float16* wl = wt + l * LW;
        transpose_cvt<<<dim3(24, 24), blk, 0, stream>>>(wq + l * WDD, wl,            DMODEL, DMODEL);
        transpose_cvt<<<dim3(24, 24), blk, 0, stream>>>(wk + l * WDD, wl + WDD,      DMODEL, DMODEL);
        transpose_cvt<<<dim3(24, 24), blk, 0, stream>>>(wv + l * WDD, wl + 2 * WDD,  DMODEL, DMODEL);
        transpose_cvt<<<dim3(24, 24), blk, 0, stream>>>(wo + l * WDD, wl + 3 * WDD,  DMODEL, DMODEL);
        transpose_cvt<<<dim3(96, 24), blk, 0, stream>>>(w1 + l * WDF, wl + 4 * WDD,  DMODEL, DFF);
        transpose_cvt<<<dim3(24, 96), blk, 0, stream>>>(w2 + l * WDF, wl + 4 * WDD + WDF, DFF, DMODEL);
    }

    embed_kernel<<<dim3((unsigned)((ND + 255) / 256)), blk, 0, stream>>>(
        inputs, emb, pos, x, x16);

    for (int l = 0; l < NLAY; ++l) {
        _Float16* wqT = wt + l * LW;
        _Float16* wkT = wqT + WDD;
        _Float16* wvT = wqT + 2 * WDD;
        _Float16* woT = wqT + 3 * WDD;
        _Float16* w1T = wqT + 4 * WDD;
        _Float16* w2T = w1T + WDF;
        float* attn_l = attns + (long long)l * ATTN_L;

        // q16/k16 = x16 @ Wq/Wk   [4096x768]x[768x768] -> f16 [B,S,H*DK]
        gemm_wmma<0><<<dim3(12, 32, 1), blk, 0, stream>>>(
            x16, 0, 0, DMODEL,  wqT, 0, 0, DMODEL,
            nullptr, q16, 0, 0, DMODEL, 1,  DMODEL, 1, nullptr, 0);
        gemm_wmma<0><<<dim3(12, 32, 1), blk, 0, stream>>>(
            x16, 0, 0, DMODEL,  wkT, 0, 0, DMODEL,
            nullptr, k16, 0, 0, DMODEL, 1,  DMODEL, 1, nullptr, 0);
        // v16t = x16 @ Wv, scattered to [B,H,DK,S] (batched over b)
        gemm_wmma<0><<<dim3(12, 16, BB), blk, 0, stream>>>(
            x16, (long long)SS * DMODEL, 0, DMODEL,  wvT, 0, 0, DMODEL,
            nullptr, v16t, (long long)NHEADS * DK * SS, 0, 1, SS,
            DMODEL, 1, nullptr, 0);

        // raw scores[b,h] = Q (SxDK) @ K^T ; k16 is naturally [n=s'][k=d]
        gemm_wmma<0><<<dim3(SS / 64, SS / 128, BB * NHEADS), blk, 0, stream>>>(
            q16, (long long)SS * DMODEL, DK, DMODEL,
            k16, (long long)SS * DMODEL, DK, DMODEL,
            attn_l, nullptr, (long long)NHEADS * SS * SS, (long long)SS * SS, SS, 1,
            DK, NHEADS, nullptr, 0);

        softmax_mask<<<dim3(SS, NHEADS, BB), blk, 0, stream>>>(attn_l, amask, 0.125f);

        // ctx16[b,h] = attn (SxS, f32) @ V ; v16t is [n=d][k=s']
        gemm_wmma<1><<<dim3(1, SS / 128, BB * NHEADS), blk, 0, stream>>>(
            attn_l, (long long)NHEADS * SS * SS, (long long)SS * SS, SS,
            v16t,   (long long)NHEADS * DK * SS, (long long)DK * SS, SS,
            nullptr, c16, (long long)SS * DMODEL, DK, DMODEL, 1,
            SS, NHEADS, nullptr, 0);

        // tmp = ctx16 @ Wo (f32 out)
        gemm_wmma<0><<<dim3(12, 32, 1), blk, 0, stream>>>(
            c16, 0, 0, DMODEL,  woT, 0, 0, DMODEL,
            tmp, nullptr, 0, 0, DMODEL, 1,  DMODEL, 1, nullptr, 0);

        add_ln<<<dim3(BB * SS), blk, 0, stream>>>(tmp, x, x16,
            ln1g + (long long)l * DMODEL, ln1b + (long long)l * DMODEL);

        // ffh16 = relu(x16 @ W1 + b1)
        gemm_wmma<0><<<dim3(DFF / 64, 32, 1), blk, 0, stream>>>(
            x16, 0, 0, DMODEL,  w1T, 0, 0, DMODEL,
            nullptr, ffh16, 0, 0, DFF, 1,  DMODEL, 1, b1 + (long long)l * DFF, 1);

        // tmp = ffh16 @ W2 + b2 (f32 out)
        gemm_wmma<0><<<dim3(12, 32, 1), blk, 0, stream>>>(
            ffh16, 0, 0, DFF,  w2T, 0, 0, DFF,
            tmp, nullptr, 0, 0, DMODEL, 1,  DFF, 1, b2 + (long long)l * DMODEL, 0);

        add_ln<<<dim3(BB * SS), blk, 0, stream>>>(tmp, x, x16,
            ln2g + (long long)l * DMODEL, ln2b + (long long)l * DMODEL);
    }

    copy_kernel<<<dim3((unsigned)((ND + 255) / 256)), blk, 0, stream>>>(x, out_x, ND);
}
